// NormalModel_30253749633309
// MI455X (gfx1250) — compile-verified, matrix operand order checked by reading, not run on hardware
//
#include <hip/hip_runtime.h>
#include <hip/hip_bf16.h>
#include <math.h>

typedef __bf16 bf16;
typedef bf16  v16bf __attribute__((ext_vector_type(16)));
typedef bf16  v8bf  __attribute__((ext_vector_type(8)));
typedef float v8f   __attribute__((ext_vector_type(8)));

#define B_    4
#define NPTS  8192
#define NH    4096            // NPTS/2
#define NTOT  12288           // NH + NPTS
#define P_TOT 32768           // B_*NPTS

static __device__ __forceinline__ bf16 to_bf(float f) { return (bf16)f; }

// -----------------------------------------------------------------------------
// WMMA GEMM:  Y[p*D + d] = sum_c X[p*Cpad + c] * W[d*Cpad + c] + bias[d]
// X: [P_TOT, Cpad] bf16 (point-major), W: [Dpad32, Cpad] bf16, Y: [P_TOT, D] f32
// One wave computes a 32(M) x 32(N) macro-tile = 2x2 16x16 subtiles:
// 4 WMMAs per 32-K step sharing 2 A-fragments and 2 B-fragments (8 b128 loads).
// A-fragment layout (16-bit A 16x32): lanes 0-15 -> M=lane, K in {0..7,16..23};
// lanes 16-31 -> M=lane-16, K in {8..15,24..31}. B mirrors with N in place of M.
// C/D layout: lane<16: N=lane, M=r; lane>=16: N=lane-16, M=r+8.
// -----------------------------------------------------------------------------
__global__ __launch_bounds__(128)
void wmma_gemm(const bf16* __restrict__ X, const bf16* __restrict__ W,
               const float* __restrict__ bias, float* __restrict__ Y,
               int Cpad, int D) {
  const int lane = threadIdx.x & 31;
  const int wave = threadIdx.x >> 5;
  const int m0 = blockIdx.x * 32;                 // 2 M subtiles
  const int n0 = (blockIdx.y * 4 + wave) * 32;    // 2 N subtiles
  const int half = lane >> 4;
  const int l15  = lane & 15;
  const int koff = half << 3;

  const bf16* xrow0 = X + (size_t)(n0 + l15) * Cpad;
  const bf16* xrow1 = X + (size_t)(n0 + 16 + l15) * Cpad;
  const bf16* wrow0 = W + (size_t)(m0 + l15) * Cpad;
  const bf16* wrow1 = W + (size_t)(m0 + 16 + l15) * Cpad;

  v8f acc00 = {}; v8f acc01 = {}; v8f acc10 = {}; v8f acc11 = {};
  for (int k = 0; k < Cpad; k += 32) {
    const int kb = k + koff;
    v8bf b0l = *(const v8bf*)(xrow0 + kb);
    v8bf b0h = *(const v8bf*)(xrow0 + kb + 16);
    v8bf b1l = *(const v8bf*)(xrow1 + kb);
    v8bf b1h = *(const v8bf*)(xrow1 + kb + 16);
    v8bf a0l = *(const v8bf*)(wrow0 + kb);
    v8bf a0h = *(const v8bf*)(wrow0 + kb + 16);
    v8bf a1l = *(const v8bf*)(wrow1 + kb);
    v8bf a1h = *(const v8bf*)(wrow1 + kb + 16);
    v16bf bb0, bb1, aa0, aa1;
#pragma unroll
    for (int i = 0; i < 8; ++i) {
      bb0[i] = b0l[i]; bb0[i + 8] = b0h[i];
      bb1[i] = b1l[i]; bb1[i + 8] = b1h[i];
      aa0[i] = a0l[i]; aa0[i + 8] = a0h[i];
      aa1[i] = a1l[i]; aa1[i + 8] = a1h[i];
    }
    acc00 = __builtin_amdgcn_wmma_f32_16x16x32_bf16(false, aa0, false, bb0,
                                                    (short)0, acc00, false, false);
    acc01 = __builtin_amdgcn_wmma_f32_16x16x32_bf16(false, aa0, false, bb1,
                                                    (short)0, acc01, false, false);
    acc10 = __builtin_amdgcn_wmma_f32_16x16x32_bf16(false, aa1, false, bb0,
                                                    (short)0, acc10, false, false);
    acc11 = __builtin_amdgcn_wmma_f32_16x16x32_bf16(false, aa1, false, bb1,
                                                    (short)0, acc11, false, false);
  }

  const int dA = m0 + (half << 3);        // M subtile 0 column base
  const int dB = m0 + 16 + (half << 3);   // M subtile 1 column base
  {
    float* yrow = Y + (size_t)(n0 + l15) * D;       // N subtile 0
#pragma unroll
    for (int r = 0; r < 8; ++r) {
      int d = dA + r; if (d < D) yrow[d] = acc00[r] + bias[d];
    }
#pragma unroll
    for (int r = 0; r < 8; ++r) {
      int d = dB + r; if (d < D) yrow[d] = acc10[r] + bias[d];
    }
  }
  {
    float* yrow = Y + (size_t)(n0 + 16 + l15) * D;  // N subtile 1
#pragma unroll
    for (int r = 0; r < 8; ++r) {
      int d = dA + r; if (d < D) yrow[d] = acc01[r] + bias[d];
    }
#pragma unroll
    for (int r = 0; r < 8; ++r) {
      int d = dB + r; if (d < D) yrow[d] = acc11[r] + bias[d];
    }
  }
}

// ---------------- BatchNorm (training stats over all P_TOT points) ----------
__global__ void zero_f32(float* p, int n) {
  int i = blockIdx.x * blockDim.x + threadIdx.x;
  if (i < n) p[i] = 0.f;
}

__global__ __launch_bounds__(256)
void bn_stats(const float* __restrict__ Y, float* __restrict__ sum,
              float* __restrict__ sumsq, int D, int rowsPerBlock) {
  const int r0 = blockIdx.x * rowsPerBlock;
  for (int c = threadIdx.x; c < D; c += blockDim.x) {
    float s = 0.f, s2 = 0.f;
    for (int r = 0; r < rowsPerBlock; ++r) {
      float v = Y[(size_t)(r0 + r) * D + c];
      s += v; s2 += v * v;
    }
    atomicAdd(&sum[c], s);
    atomicAdd(&sumsq[c], s2);
  }
}

__global__ void bn_final(const float* __restrict__ sum, const float* __restrict__ sumsq,
                         const float* __restrict__ g, const float* __restrict__ b,
                         float* __restrict__ scl, float* __restrict__ sft,
                         int D, float invP) {
  int d = blockIdx.x * blockDim.x + threadIdx.x;
  if (d >= D) return;
  float m   = sum[d] * invP;
  float var = sumsq[d] * invP - m * m;
  float a   = g[d] * rsqrtf(var + 1e-5f);
  scl[d] = a;
  sft[d] = b[d] - m * a;
}

__global__ __launch_bounds__(256)
void bn_apply(float* __restrict__ Y, const float* __restrict__ scl,
              const float* __restrict__ sft, bf16* __restrict__ Xout,
              int D, int relu, size_t total) {
  size_t i = (size_t)blockIdx.x * blockDim.x + threadIdx.x;
  if (i >= total) return;
  int d = (int)(i % (size_t)D);
  float v = Y[i] * scl[d] + sft[d];
  if (relu) v = fmaxf(v, 0.f);
  Y[i] = v;
  if (Xout) Xout[i] = to_bf(v);
}

// ---------------- max over points per (batch, channel) -----------------------
__global__ void colmax(const float* __restrict__ Y, float* __restrict__ out, int D) {
  int idx = blockIdx.x * blockDim.x + threadIdx.x;  // b*D + d
  if (idx >= B_ * D) return;
  int b = idx / D, d = idx % D;
  const float* base = Y + (size_t)b * NPTS * D + d;
  float m = -1e30f;
  for (int n = 0; n < NPTS; ++n) m = fmaxf(m, base[(size_t)n * D]);
  out[idx] = m;
}

// ---------------- small FC layers (M = 4) ------------------------------------
__global__ void fc(const float* __restrict__ X, const float* __restrict__ W,
                   const float* __restrict__ bias, float* __restrict__ out,
                   int I, int O, int relu) {
  int idx = blockIdx.x * blockDim.x + threadIdx.x;
  if (idx >= B_ * O) return;
  int b = idx / O, o = idx % O;
  const float* x = X + (size_t)b * I;
  const float* w = W + (size_t)o * I;
  float s = bias[o];
  for (int i = 0; i < I; ++i) s += x[i] * w[i];
  if (relu) s = fmaxf(s, 0.f);
  out[idx] = s;
}

// BN over the batch dim only (B=4), + ReLU, in place.
__global__ void bn_small(float* __restrict__ z, const float* __restrict__ g,
                         const float* __restrict__ b, int D) {
  int d = blockIdx.x * blockDim.x + threadIdx.x;
  if (d >= D) return;
  float s = 0.f, s2 = 0.f;
  for (int i = 0; i < B_; ++i) { float v = z[i * D + d]; s += v; s2 += v * v; }
  float m = s * 0.25f, var = s2 * 0.25f - m * m;
  float istd = rsqrtf(var + 1e-5f);
  for (int i = 0; i < B_; ++i) {
    float v = (z[i * D + d] - m) * istd * g[d] + b[d];
    z[i * D + d] = fmaxf(v, 0.f);
  }
}

// ---------------- data movement / conversion helpers -------------------------
__global__ void wconv(const float* __restrict__ W, bf16* __restrict__ Wb,
                      int D, int C, int Dpad, int Cpad) {
  int i = blockIdx.x * blockDim.x + threadIdx.x;
  if (i >= Dpad * Cpad) return;
  int r = i / Cpad, c = i % Cpad;
  Wb[i] = (r < D && c < C) ? to_bf(W[(size_t)r * C + c]) : to_bf(0.f);
}

__global__ void build_in32(const float* __restrict__ pts, bf16* __restrict__ X,
                           int ncomp) {
  int i = blockIdx.x * blockDim.x + threadIdx.x;  // point index in [0, P_TOT)
  if (i >= P_TOT) return;
  bf16* row = X + (size_t)i * 32;
  for (int c = 0; c < 32; ++c) row[c] = to_bf(0.f);
  for (int c = 0; c < ncomp; ++c) row[c] = to_bf(pts[(size_t)i * ncomp + c]);
}

__global__ void f2bf(const float* __restrict__ in, bf16* __restrict__ out, int n) {
  int i = blockIdx.x * blockDim.x + threadIdx.x;
  if (i < n) out[i] = to_bf(in[i]);
}

// mpc [B, NTOT, 4] = concat( [out_hole, flag=0], [in_partial, flag=1] )
__global__ void build_mpc(const float* __restrict__ hole, const float* __restrict__ part,
                          float* __restrict__ mpc) {
  int i = blockIdx.x * blockDim.x + threadIdx.x;
  if (i >= B_ * NTOT) return;
  int b = i / NTOT, n = i % NTOT;
  float x, y, z, f;
  if (n < NH) {
    const float* p = hole + ((size_t)b * NH + n) * 3;
    x = p[0]; y = p[1]; z = p[2]; f = 0.f;
  } else {
    const float* p = part + ((size_t)b * NPTS + (n - NH)) * 3;
    x = p[0]; y = p[1]; z = p[2]; f = 1.f;
  }
  float* o = mpc + (size_t)i * 4;
  o[0] = x; o[1] = y; o[2] = z; o[3] = f;
}

// farthest point sampling: one workgroup per batch, dist[] lives in LDS
__global__ __launch_bounds__(1024)
void fps_kernel(const float* __restrict__ mpc, int* __restrict__ idx) {
  __shared__ float dist[NTOT];
  __shared__ float rmax[1024];
  __shared__ int   rarg[1024];
  __shared__ int   sfar;
  const int b = blockIdx.x;
  const float* pts = mpc + (size_t)b * NTOT * 4;
  const int t = threadIdx.x;
  for (int p = t; p < NTOT; p += 1024) dist[p] = 1e10f;
  if (t == 0) sfar = 1;  // reference starts at index 1
  __syncthreads();
  for (int it = 0; it < NPTS; ++it) {
    const int far = sfar;
    if (t == 0) idx[b * NPTS + it] = far;
    const float cx = pts[far * 4 + 0], cy = pts[far * 4 + 1], cz = pts[far * 4 + 2];
    float bm = -1.f; int ba = 0;
    for (int p = t; p < NTOT; p += 1024) {
      const float* q = pts + (size_t)p * 4;
      float dx = q[0] - cx, dy = q[1] - cy, dz = q[2] - cz;
      float dd = fminf(dist[p], dx * dx + dy * dy + dz * dz);
      dist[p] = dd;
      if (dd > bm) { bm = dd; ba = p; }
    }
    rmax[t] = bm; rarg[t] = ba;
    __syncthreads();
    for (int s = 512; s > 0; s >>= 1) {
      if (t < s && rmax[t + s] > rmax[t]) { rmax[t] = rmax[t + s]; rarg[t] = rarg[t + s]; }
      __syncthreads();
    }
    if (t == 0) sfar = rarg[0];
    __syncthreads();
  }
}

__global__ void gather_spc(const float* __restrict__ mpc, const int* __restrict__ idx,
                           float* __restrict__ spc) {
  int i = blockIdx.x * blockDim.x + threadIdx.x;  // b*NPTS + n
  if (i >= B_ * NPTS) return;
  int b = i / NPTS;
  int j = idx[i];
  const float* s = mpc + ((size_t)b * NTOT + j) * 4;
  float* o = spc + (size_t)i * 4;
  o[0] = s[0]; o[1] = s[1]; o[2] = s[2]; o[3] = s[3];
}

// hcat [P, 1088] = [ broadcast gmax[b, 0:1024] , pointfeat[p, 0:64] ]
__global__ __launch_bounds__(256)
void build_hcat(const bf16* __restrict__ gmax_bf, const bf16* __restrict__ pf,
                bf16* __restrict__ X) {
  size_t i = (size_t)blockIdx.x * blockDim.x + threadIdx.x;
  const size_t total = (size_t)P_TOT * 1088;
  if (i >= total) return;
  int p = (int)(i / 1088), c = (int)(i % 1088);
  X[i] = (c < 1024) ? gmax_bf[(p / NPTS) * 1024 + c]
                    : pf[(size_t)p * 64 + (c - 1024)];
}

// out_complete = spc.xyz + tanh(conv_out)*0.15
__global__ void finalize(const float* __restrict__ spc, const float* __restrict__ Y3,
                         float* __restrict__ outc) {
  int i = blockIdx.x * blockDim.x + threadIdx.x;  // b*NPTS + n
  if (i >= P_TOT) return;
  const float* s = spc + (size_t)i * 4;
  const float* y = Y3 + (size_t)i * 3;
  float* o = outc + (size_t)i * 3;
  for (int k = 0; k < 3; ++k) o[k] = s[k] + tanhf(y[k]) * 0.15f;
}

// =============================================================================
extern "C" void kernel_launch(void* const* d_in, const int* in_sizes, int n_in,
                              void* d_out, int out_size, void* d_ws, size_t ws_size,
                              hipStream_t stream) {
  (void)in_sizes; (void)n_in; (void)out_size; (void)ws_size;
  // inputs: 0 in_partial, 1 in_hole, 2 in_complete, then params (sorted keys):
  //  0 d1_b 1 d1_w 2 d2_b 3 d2_w 4 d3_b 5 d3_w
  //  6 e_bn1_b 7 e_bn1_g 8 e_bn2_b 9 e_bn2_g 10 e_bn3_b 11 e_bn3_g 12 e_bnl_b 13 e_bnl_g
  // 14 e_c1_b 15 e_c1_w 16 e_c2_b 17 e_c2_w 18 e_c3_b 19 e_c3_w 20 e_lin_b 21 e_lin_w
  // 22 r_bn1_b 23 r_bn1_g 24 r_bn2_b 25 r_bn2_g 26 r_bn3_b 27 r_bn3_g
  // 28 r_bn4_b 29 r_bn4_g 30 r_bn5_b 31 r_bn5_g 32 r_bn6_b 33 r_bn6_g
  // 34 r_c1_b 35 r_c1_w 36 r_c2_b 37 r_c2_w 38 r_c3_b 39 r_c3_w 40 r_c4_b 41 r_c4_w
  // 42 r_c5_b 43 r_c5_w 44 r_c6_b 45 r_c6_w 46 r_c7_b 47 r_c7_w
#define PRM(i) ((const float*)d_in[3 + (i)])
  const float* in_partial = (const float*)d_in[0];
  float* out_hole     = (float*)d_out;                  // [B, NH, 3]
  float* out_complete = (float*)d_out + (size_t)B_ * NH * 3;

  // ---- workspace layout ----
  char* ws = (char*)d_ws;
  size_t off = 0;
  auto alloc = [&](size_t bytes) { size_t o = off; off += (bytes + 255) & ~(size_t)255; return o; };
  float* Yf   = (float*)(ws + alloc((size_t)P_TOT * 1024 * 4));
  bf16*  Xbf  = (bf16*)(ws + alloc((size_t)P_TOT * 1088 * 2));
  bf16*  Wbf  = (bf16*)(ws + alloc((size_t)1100000 * 2));
  float* sum  = (float*)(ws + alloc(1088 * 4));
  float* sq   = (float*)(ws + alloc(1088 * 4));
  float* scl  = (float*)(ws + alloc(1088 * 4));
  float* sft  = (float*)(ws + alloc(1088 * 4));
  float* gfeat = (float*)(ws + alloc(4096 * 4));
  float* zb    = (float*)(ws + alloc(4096 * 4));
  float* h1    = (float*)(ws + alloc(4096 * 4));
  float* h2    = (float*)(ws + alloc(4096 * 4));
  float* gmax  = (float*)(ws + alloc(4096 * 4));
  bf16*  gmax_bf = (bf16*)(ws + alloc(4096 * 2));
  bf16*  pf_bf   = (bf16*)(ws + alloc((size_t)P_TOT * 64 * 2));
  float* mpc  = (float*)(ws + alloc((size_t)B_ * NTOT * 4 * 4));
  float* spc  = (float*)(ws + alloc((size_t)B_ * NPTS * 4 * 4));
  int*   fidx = (int*)(ws + alloc((size_t)B_ * NPTS * 4));

  auto cdiv = [](size_t a, size_t b) { return (int)((a + b - 1) / b); };
  size_t woff = 0;

  // one conv layer: convert weights, run WMMA GEMM into Yf
  auto conv_layer = [&](const bf16* Xin, int Cpad, int widx, int bidx, int D, int C) {
    int Dpad = (D + 31) & ~31;
    bf16* wb = Wbf + woff;
    woff += (size_t)Dpad * Cpad;
    wconv<<<cdiv((size_t)Dpad * Cpad, 256), 256, 0, stream>>>(PRM(widx), wb, D, C, Dpad, Cpad);
    dim3 g(Dpad / 32, P_TOT / 128);   // 4 waves/block, each 32x32 macro-tile
    wmma_gemm<<<g, 128, 0, stream>>>(Xin, wb, PRM(bidx), Yf, Cpad, D);
  };
  // BN over all points (+optional ReLU, +optional bf16 export for next layer)
  auto bn_layer = [&](int gidx, int bidx, int D, int relu, bf16* Xout) {
    zero_f32<<<cdiv(D, 256), 256, 0, stream>>>(sum, D);
    zero_f32<<<cdiv(D, 256), 256, 0, stream>>>(sq, D);
    bn_stats<<<64, 256, 0, stream>>>(Yf, sum, sq, D, P_TOT / 64);
    bn_final<<<cdiv(D, 256), 256, 0, stream>>>(sum, sq, PRM(gidx), PRM(bidx), scl, sft,
                                               D, 1.0f / (float)P_TOT);
    size_t total = (size_t)P_TOT * D;
    bn_apply<<<cdiv(total, 256), 256, 0, stream>>>(Yf, scl, sft, Xout, D, relu, total);
  };

  // ---------------- encoder ----------------
  build_in32<<<cdiv(P_TOT, 256), 256, 0, stream>>>(in_partial, Xbf, 3);
  conv_layer(Xbf, 32, 15, 14, 64, 3);          // e_c1
  bn_layer(7, 6, 64, 1, Xbf);
  conv_layer(Xbf, 64, 17, 16, 128, 64);        // e_c2
  bn_layer(9, 8, 128, 1, Xbf);
  conv_layer(Xbf, 128, 19, 18, 1024, 128);     // e_c3
  bn_layer(11, 10, 1024, 0, (bf16*)nullptr);
  colmax<<<cdiv((size_t)B_ * 1024, 256), 256, 0, stream>>>(Yf, gfeat, 1024);
  fc<<<cdiv((size_t)B_ * 1024, 256), 256, 0, stream>>>(gfeat, PRM(21), PRM(20), zb, 1024, 1024, 0);
  bn_small<<<cdiv(1024, 256), 256, 0, stream>>>(zb, PRM(13), PRM(12), 1024);

  // ---------------- decoder ----------------
  fc<<<cdiv((size_t)B_ * 1024, 256), 256, 0, stream>>>(zb, PRM(1), PRM(0), h1, 1024, 1024, 1);
  fc<<<cdiv((size_t)B_ * 1024, 256), 256, 0, stream>>>(h1, PRM(3), PRM(2), h2, 1024, 1024, 1);
  fc<<<cdiv((size_t)B_ * NH * 3, 256), 256, 0, stream>>>(h2, PRM(5), PRM(4), out_hole,
                                                         1024, NH * 3, 0);

  // ---------------- merge + FPS + gather ----------------
  build_mpc<<<cdiv((size_t)B_ * NTOT, 256), 256, 0, stream>>>(out_hole, in_partial, mpc);
  fps_kernel<<<B_, 1024, 0, stream>>>(mpc, fidx);
  gather_spc<<<cdiv((size_t)B_ * NPTS, 256), 256, 0, stream>>>(mpc, fidx, spc);

  // ---------------- refinement ----------------
  build_in32<<<cdiv(P_TOT, 256), 256, 0, stream>>>(spc, Xbf, 4);
  conv_layer(Xbf, 32, 35, 34, 64, 4);          // r_c1
  bn_layer(23, 22, 64, 1, pf_bf);              // pointfeat
  conv_layer(pf_bf, 64, 37, 36, 128, 64);      // r_c2
  bn_layer(25, 24, 128, 1, Xbf);
  conv_layer(Xbf, 128, 39, 38, 1024, 128);     // r_c3
  bn_layer(27, 26, 1024, 0, (bf16*)nullptr);
  colmax<<<cdiv((size_t)B_ * 1024, 256), 256, 0, stream>>>(Yf, gmax, 1024);
  f2bf<<<cdiv((size_t)B_ * 1024, 256), 256, 0, stream>>>(gmax, gmax_bf, B_ * 1024);
  build_hcat<<<cdiv((size_t)P_TOT * 1088, 256), 256, 0, stream>>>(gmax_bf, pf_bf, Xbf);
  conv_layer(Xbf, 1088, 41, 40, 512, 1088);    // r_c4
  bn_layer(29, 28, 512, 1, Xbf);
  conv_layer(Xbf, 512, 43, 42, 256, 512);      // r_c5
  bn_layer(31, 30, 256, 1, Xbf);
  conv_layer(Xbf, 256, 45, 44, 128, 256);      // r_c6
  bn_layer(33, 32, 128, 1, Xbf);
  conv_layer(Xbf, 128, 47, 46, 3, 128);        // r_c7 (no BN)
  finalize<<<cdiv(P_TOT, 256), 256, 0, stream>>>(spc, Yf, out_complete);
#undef PRM
}
